// FrameTransformer_6811818131912
// MI455X (gfx1250) — matrix operand order repaired and checked by reading, not compiled, not run on hardware
//
#include <hip/hip_runtime.h>
#include <hip/hip_bf16.h>

// FrameTransformer forward for MI455X (gfx1250, wave32, WMMA + TDM).
// Big GEMMs + attention run on v_wmma_f32_16x16x32_bf16 (fp32 accum).
// GEMM A-tiles are staged into LDS by the Tensor Data Mover (TENSORcnt),
// using TDM's pad feature to produce the 80B padded LDS row pitch.
// Workspace requirement: ~529 MB (see offsets in kernel_launch).

typedef __bf16 bf16_t;
typedef __attribute__((ext_vector_type(16))) __bf16 v16bf;
typedef __attribute__((ext_vector_type(8)))  __bf16 v8bf;
typedef __attribute__((ext_vector_type(4)))  __bf16 v4bf;
typedef __attribute__((ext_vector_type(8)))  float  v8f;

typedef unsigned int u32x4 __attribute__((ext_vector_type(4)));
typedef int          i32x4 __attribute__((ext_vector_type(4)));
typedef int          i32x8 __attribute__((ext_vector_type(8)));

#if defined(__gfx1250__) && __has_builtin(__builtin_amdgcn_tensor_load_to_lds)
#define FT_HAVE_TDM 1
#else
#define FT_HAVE_TDM 0
#endif

#define BDIM 2
#define CH 12
#define FEATD 1024
#define WDIM 512
#define HEADS 4
#define HDIM 256

__device__ __forceinline__ v16bf frag_ld(const bf16_t* p0, const bf16_t* p1) {
  // Two contiguous 16-byte loads -> one 16-element bf16 fragment.
  v8bf lo = *(const v8bf*)p0;
  v8bf hi = *(const v8bf*)p1;
  return __builtin_shufflevector(lo, hi, 0,1,2,3,4,5,6,7,8,9,10,11,12,13,14,15);
}

__device__ __forceinline__ v8f zero8() {
  v8f z = {0.f,0.f,0.f,0.f,0.f,0.f,0.f,0.f};
  return z;
}

// ---------------------------------------------------------------- embed 1x1
__global__ void k_embed(const float* __restrict__ x, const float* __restrict__ ew,
                        const float* __restrict__ eb, float* __restrict__ h0) {
  size_t idx = (size_t)blockIdx.x * blockDim.x + threadIdx.x;
  const size_t plane = (size_t)FEATD * WDIM;
  const size_t total = (size_t)BDIM * CH * plane;
  if (idx >= total) return;
  size_t fw = idx % plane;
  int oc = (int)((idx / plane) % CH);
  int b  = (int)(idx / (plane * CH));
  const float* xb = x + (size_t)b * 3 * plane + fw;
  float a = eb[oc];
  a += xb[0]         * ew[oc * 3 + 0];
  a += xb[plane]     * ew[oc * 3 + 1];
  a += xb[2 * plane] * ew[oc * 3 + 2];
  h0[idx] = a;
}

__global__ void k_copy_x(const float* __restrict__ x, float* __restrict__ out0) {
  size_t idx = (size_t)blockIdx.x * blockDim.x + threadIdx.x;
  const size_t plane = (size_t)FEATD * WDIM;
  const size_t total = (size_t)BDIM * 3 * plane;
  if (idx >= total) return;
  size_t fw = idx % plane;
  int ci = (int)((idx / plane) % 3);
  int b  = (int)(idx / (plane * 3));
  out0[((size_t)(b * 15 + ci)) * plane + fw] = x[idx];
}

// ------------------------------------------------------- layernorm over FEAT
__global__ __launch_bounds__(256) void k_layernorm(
    const float* __restrict__ src, float* __restrict__ dst,
    const float* __restrict__ g, const float* __restrict__ bt) {
  __shared__ float s1[256], s2[256];
  const int col = blockIdx.x;            // b*c*w columns
  const int bc = col / WDIM;
  const int w  = col % WDIM;
  const int c  = bc % CH;
  const int tid = threadIdx.x;
  float v[4]; float su = 0.f, sq = 0.f;
#pragma unroll
  for (int j = 0; j < 4; ++j) {
    int f = tid + j * 256;
    v[j] = src[((size_t)bc * FEATD + f) * WDIM + w];
    su += v[j]; sq += v[j] * v[j];
  }
  s1[tid] = su; s2[tid] = sq; __syncthreads();
  for (int s = 128; s > 0; s >>= 1) {
    if (tid < s) { s1[tid] += s1[tid + s]; s2[tid] += s2[tid + s]; }
    __syncthreads();
  }
  float mean = s1[0] * (1.0f / FEATD);
  float var  = s2[0] * (1.0f / FEATD) - mean * mean;
  float rstd = rsqrtf(var + 1e-8f);
#pragma unroll
  for (int j = 0; j < 4; ++j) {
    int f = tid + j * 256;
    dst[((size_t)bc * FEATD + f) * WDIM + w] =
        (v[j] - mean) * rstd * g[c * FEATD + f] + bt[c * FEATD + f];
  }
}

// -------------------------------------------------------- depthwise 9x9 conv
__global__ __launch_bounds__(256) void k_dwconv9(
    const float* __restrict__ src, const float* __restrict__ w9,
    bf16_t* __restrict__ dst) {
  __shared__ float sIn[24 * 24];
  __shared__ float sW[81];
  const int bc = blockIdx.z;
  const int h0 = blockIdx.y * 16;
  const int w0 = blockIdx.x * 16;
  const int tid = threadIdx.x;
  const float* sp = src + (size_t)bc * FEATD * WDIM;
  const float* wp = w9 + (size_t)(bc % CH) * 81;
  if (tid < 81) sW[tid] = wp[tid];
  for (int li = tid; li < 576; li += 256) {
    int iy = li / 24, ix = li % 24;
    int h = h0 + iy - 4, w = w0 + ix - 4;
    float v = 0.f;
    if (h >= 0 && h < FEATD && w >= 0 && w < WDIM) v = sp[(size_t)h * WDIM + w];
    sIn[li] = v;
  }
  __syncthreads();
  int ty = tid >> 4, tx = tid & 15;
  float acc = 0.f;
#pragma unroll
  for (int ky = 0; ky < 9; ++ky)
#pragma unroll
    for (int kx = 0; kx < 9; ++kx)
      acc += sIn[(ty + ky) * 24 + tx + kx] * sW[ky * 9 + kx];
  dst[((size_t)bc * FEATD + h0 + ty) * WDIM + w0 + tx] = (bf16_t)acc;
}

// ----------------------------------------------------------- fp32 -> bf16
__global__ void k_f32_to_bf16(const float* __restrict__ s, bf16_t* __restrict__ d,
                              long long n) {
  long long i = (long long)blockIdx.x * blockDim.x + threadIdx.x;
  if (i < n) d[i] = (bf16_t)s[i];
}

// ------------------------------------------------ batched WMMA GEMM (bf16)
// C[z] (MxN, f32) = A[z % aMod] (MxK, bf16 row-major) * B[z] (KxN, bf16 row-major)
// block tile 64M x 32N, 8 waves (4x2 of 16x16), K-step 32.
// A tile (64x32, 80B padded LDS pitch) staged by the Tensor Data Mover
// (one tensor_load_to_lds per workgroup per K-step, wave 0 issues, TENSORcnt
// waited before the workgroup barrier). B tile transposed through VGPRs.
__global__ __launch_bounds__(256) void k_gemm_bf16(
    const bf16_t* __restrict__ A, const bf16_t* __restrict__ B,
    float* __restrict__ C, int M, int N, int K, long long aStride, int aMod) {
  extern __shared__ __align__(16) char smem[];
  bf16_t* sA = (bf16_t*)smem;             // [64][40] at dynamic-LDS offset 0
  bf16_t* sB = (bf16_t*)(smem + 5120);    // [32][40] (transposed), pad to 40
  const int z = blockIdx.z;
  const bf16_t* Ab = A + (long long)(z % aMod) * aStride;
  const bf16_t* Bb = B + (long long)z * K * N;
  float* Cb = C + (long long)z * M * N;
  const int tid = threadIdx.x;
  const int lane = tid & 31, wave = tid >> 5;
  const int mi = wave >> 1, ni = wave & 1;
  const int m0 = blockIdx.y * 64, n0 = blockIdx.x * 32;
  const bool lo = lane < 16;
  const int l15 = lane & 15;
  const int ab = lo ? 0 : 8;       // A fragment K base (ISA 16-bit A layout)
  const int bb = lo ? 0 : 8;       // B fragment K base (ISA 16-bit B layout)
  const int bk = tid >> 3, bn = (tid & 7) * 4;   // B stage: 4 bf16/thread
  v8f acc = zero8();
  for (int k0 = 0; k0 < K; k0 += 32) {
    __syncthreads();
#if FT_HAVE_TDM
    if (wave == 0) {
      // D# group0: count=1, lds_addr=0, 57-bit global addr, type=2 (image)
      unsigned long long ga =
          (unsigned long long)(const void*)(Ab + (size_t)m0 * K + k0);
      u32x4 g0 = { 1u, 0u, (unsigned)(ga & 0xffffffffu),
                   ((unsigned)((ga >> 32) & 0x01ffffffu)) | (2u << 30) };
      // D# group1: data_size=2B, pad every 16 DWORDs (64B row) by 4 DWORDs
      // (16B) -> LDS pitch 80B; tensor K x M, tile 32x64, dim0 stride = K.
      i32x8 g1 = {
        (int)((1u << 16) | (1u << 20) | (3u << 22) | (3u << 25)),
        (int)(((unsigned)K & 0xffffu) << 16),                 // dim0 lo16
        (int)((((unsigned)K >> 16) & 0xffffu) |
              (((unsigned)M & 0xffffu) << 16)),               // dim0 hi|dim1 lo
        (int)((((unsigned)M >> 16) & 0xffffu) | (32u << 16)), // dim1 hi|tile0
        (int)64,                                              // tile_dim1
        (int)(unsigned)K,                                     // dim0 stride lo
        0, 0 };
      i32x4 gz = {0, 0, 0, 0};
#if __clang_major__ >= 23
      i32x8 gz8 = {0, 0, 0, 0, 0, 0, 0, 0};
      __builtin_amdgcn_tensor_load_to_lds(g0, g1, gz, gz, gz8, 0);
#else
      __builtin_amdgcn_tensor_load_to_lds(g0, g1, gz, gz, 0);
#endif
    }
#else
    {
      const int am = tid >> 2, ak = (tid & 3) * 8;  // A stage: 8 bf16/thread
      *(v8bf*)&sA[am * 40 + ak] =
          *(const v8bf*)&Ab[(size_t)(m0 + am) * K + k0 + ak];
    }
#endif
    // B transpose staging (all waves)
    v4bf bv = *(const v4bf*)&Bb[(size_t)(k0 + bk) * N + n0 + bn];
#pragma unroll
    for (int i = 0; i < 4; ++i) sB[(bn + i) * 40 + bk] = bv[i];
    if (k0 + 32 < K)  // prefetch next B K-block (global_prefetch_b8)
      __builtin_prefetch(&Bb[(size_t)(k0 + 32 + bk) * N + n0 + bn], 0, 1);
#if FT_HAVE_TDM
    if (wave == 0) {
#if __has_builtin(__builtin_amdgcn_s_wait_tensorcnt)
      __builtin_amdgcn_s_wait_tensorcnt((short)0);
#else
      asm volatile("s_wait_tensorcnt 0x0" ::: "memory");
#endif
    }
#endif
    __syncthreads();
    v16bf afr = frag_ld(&sA[(mi * 16 + l15) * 40 + ab],
                        &sA[(mi * 16 + l15) * 40 + ab + 16]);
    v16bf bfr = frag_ld(&sB[(ni * 16 + l15) * 40 + bb],
                        &sB[(ni * 16 + l15) * 40 + bb + 16]);
    acc = __builtin_amdgcn_wmma_f32_16x16x32_bf16(false, afr, false, bfr,
                                                  (short)0, acc, false, false);
  }
  const int row0 = m0 + mi * 16 + (lo ? 0 : 8);
  const int col  = n0 + ni * 16 + l15;
#pragma unroll
  for (int r = 0; r < 8; ++r)
    Cb[(size_t)(row0 + r) * N + col] = acc[r];
}

// ------------------------------------------------------ rope + head packing
__global__ void k_rope_pack(const float* __restrict__ y, bf16_t* __restrict__ dst) {
  int idx = blockIdx.x * blockDim.x + threadIdx.x;   // bc:n:t:i
  if (idx >= BDIM * CH * HEADS * WDIM * (HDIM / 2)) return;
  int i  = idx & 127;
  int t  = (idx >> 7) & 511;
  int n  = (idx >> 16) & 3;
  int bc = idx >> 18;
  size_t sb = ((size_t)bc * FEATD + n * HDIM + 2 * i) * WDIM + t;
  float x0 = y[sb];
  float x1 = y[sb + WDIM];
  float inv = __expf(-((float)(2 * i) / (float)HDIM) * 9.210340371976184f); // ln 1e4
  float fr = (float)t * inv;
  float sn, cs; sincosf(fr, &sn, &cs);
  size_t db = (((size_t)bc * HEADS + n) * WDIM + t) * HDIM;
  dst[db + 2 * i]     = (bf16_t)(x0 * cs - x1 * sn);
  dst[db + 2 * i + 1] = (bf16_t)(x1 * cs + x0 * sn);
}

__global__ void k_pack_v(const float* __restrict__ y, bf16_t* __restrict__ dst) {
  int idx = blockIdx.x * blockDim.x + threadIdx.x;   // bc:n:t:d (matches dst flat)
  if (idx >= BDIM * CH * HEADS * WDIM * HDIM) return;
  int d  = idx & 255;
  int t  = (idx >> 8) & 511;
  int n  = (idx >> 17) & 3;
  int bc = idx >> 19;
  dst[idx] = (bf16_t)(y[((size_t)bc * FEATD + n * HDIM + d) * WDIM + t]);
}

// --------------------------- fused attention: S=QK^T -> qk out -> softmax -> PV
__global__ __launch_bounds__(256) void k_attention(
    const bf16_t* __restrict__ Q, const bf16_t* __restrict__ Kf,
    const bf16_t* __restrict__ V, float* __restrict__ qk_out,
    bf16_t* __restrict__ att_out) {
  __shared__ __align__(16) float  sS[16 * 512];   // scores; reused as V^T tile
  __shared__ __align__(16) bf16_t sP[16 * 520];   // softmax probs (padded rows)
  __shared__ float red[16 * 16];
  const int bcn = blockIdx.y;       // (b*12+c)*4+n
  const int m0  = blockIdx.x * 16;  // q-row tile
  const bf16_t* qp = Q + (size_t)bcn * WDIM * HDIM;
  const bf16_t* kp = Kf + (size_t)bcn * WDIM * HDIM;
  const bf16_t* vp = V + (size_t)bcn * WDIM * HDIM;
  const int tid = threadIdx.x;
  const int lane = tid & 31, wave = tid >> 5;
  const bool lo = lane < 16;
  const int l15 = lane & 15;
  const int ab = lo ? 0 : 8, bb = lo ? 0 : 8;

  // Phase 1: S = (Q K^T) / 32 ; write qk output + LDS scores
  {
    v8f accs[4] = {zero8(), zero8(), zero8(), zero8()};
    const bf16_t* qrow = qp + (size_t)(m0 + l15) * HDIM;
    const int nt0 = wave * 4;
    for (int kk = 0; kk < HDIM; kk += 32) {
      v16bf afr = frag_ld(qrow + kk + ab, qrow + kk + ab + 16);
#pragma unroll
      for (int j = 0; j < 4; ++j) {
        const bf16_t* krow = kp + (size_t)((nt0 + j) * 16 + l15) * HDIM + kk + bb;
        v16bf bfr = frag_ld(krow, krow + 16);
        accs[j] = __builtin_amdgcn_wmma_f32_16x16x32_bf16(
            false, afr, false, bfr, (short)0, accs[j], false, false);
      }
    }
    float* qb = qk_out + (size_t)bcn * WDIM * WDIM;
#pragma unroll
    for (int j = 0; j < 4; ++j) {
      int ncol = (nt0 + j) * 16 + l15;
#pragma unroll
      for (int r = 0; r < 8; ++r) {
        int mrow = r + (lo ? 0 : 8);
        float s = accs[j][r] * 0.03125f;         // 1/sqrt(1024)
        sS[mrow * 512 + ncol] = s;
        qb[(size_t)(m0 + mrow) * WDIM + ncol] = s;
      }
    }
  }
  __syncthreads();

  // Phase 2: row softmax in LDS, result as bf16 probs in sP
  {
    const int m = tid >> 4, j = tid & 15;
    float lmax = -3.4e38f;
    for (int c = j; c < 512; c += 16) lmax = fmaxf(lmax, sS[m * 512 + c]);
    red[m * 16 + j] = lmax; __syncthreads();
    for (int s = 8; s > 0; s >>= 1) {
      if (j < s) red[m * 16 + j] = fmaxf(red[m * 16 + j], red[m * 16 + j + s]);
      __syncthreads();
    }
    float rmax = red[m * 16];
    __syncthreads();
    float lsum = 0.f;
    for (int c = j; c < 512; c += 16) {
      float e = __expf(sS[m * 512 + c] - rmax);
      sS[m * 512 + c] = e; lsum += e;
    }
    red[m * 16 + j] = lsum; __syncthreads();
    for (int s = 8; s > 0; s >>= 1) {
      if (j < s) red[m * 16 + j] += red[m * 16 + j + s];
      __syncthreads();
    }
    float rinv = 1.0f / red[m * 16];
    for (int c = j; c < 512; c += 16)
      sP[m * 520 + c] = (bf16_t)(sS[m * 512 + c] * rinv);
  }
  __syncthreads();

  // Phase 3: O = P (16x512) * V (512x256); V staged transposed through LDS
  {
    bf16_t* sV = (bf16_t*)sS;                  // [d][k], row pad 40, 20 KB
    v8f oacc[2] = {zero8(), zero8()};
    const int nt0 = wave * 2;
    const int vk = tid >> 3, vd4 = (tid & 7) * 4;
    for (int kk = 0; kk < WDIM; kk += 32) {
      __syncthreads();
#pragma unroll
      for (int dblk = 0; dblk < 8; ++dblk) {
        int d = dblk * 32 + vd4;
        v4bf t = *(const v4bf*)&vp[(size_t)(kk + vk) * HDIM + d];
#pragma unroll
        for (int i = 0; i < 4; ++i) sV[(d + i) * 40 + vk] = t[i];
      }
      __syncthreads();
      v16bf afr = frag_ld(&sP[l15 * 520 + kk + ab], &sP[l15 * 520 + kk + ab + 16]);
#pragma unroll
      for (int j = 0; j < 2; ++j) {
        int dc = (nt0 + j) * 16 + l15;
        v16bf bfr = frag_ld(&sV[dc * 40 + bb], &sV[dc * 40 + bb + 16]);
        oacc[j] = __builtin_amdgcn_wmma_f32_16x16x32_bf16(
            false, afr, false, bfr, (short)0, oacc[j], false, false);
      }
    }
    const int bc = bcn >> 2, nh = bcn & 3;
#pragma unroll
    for (int j = 0; j < 2; ++j) {
      int d = (nt0 + j) * 16 + l15;
#pragma unroll
      for (int r = 0; r < 8; ++r) {
        int mrow = r + (lo ? 0 : 8);
        att_out[((size_t)bc * FEATD + nh * HDIM + d) * WDIM + (m0 + mrow)] =
            (bf16_t)oacc[j][r];
      }
    }
  }
}

// ------------------------------------------------------------- h1 = h0 + z
__global__ void k_add(const float* __restrict__ a, const float* __restrict__ b,
                      float* __restrict__ o, long long n) {
  long long i = (long long)blockIdx.x * blockDim.x + threadIdx.x;
  if (i < n) o[i] = a[i] + b[i];
}

// --------------------------------------------- conv1 3x3 (12->48) + relu^2
__global__ __launch_bounds__(256) void k_conv1(
    const float* __restrict__ src, const float* __restrict__ w,
    const float* __restrict__ bias, bf16_t* __restrict__ dst) {
  __shared__ float sIn[12 * 18 * 18];
  __shared__ float sW[48 * 12 * 9];
  const int b = blockIdx.z;
  const int h0 = blockIdx.y * 16, w0 = blockIdx.x * 16;
  const int tid = threadIdx.x;
  for (int li = tid; li < 48 * 12 * 9; li += 256) sW[li] = w[li];
  for (int li = tid; li < 12 * 324; li += 256) {
    int ci = li / 324, rem = li % 324;
    int iy = rem / 18, ix = rem % 18;
    int h = h0 + iy - 1, ww = w0 + ix - 1;
    float v = 0.f;
    if (h >= 0 && h < FEATD && ww >= 0 && ww < WDIM)
      v = src[(((size_t)b * CH + ci) * FEATD + h) * WDIM + ww];
    sIn[li] = v;
  }
  __syncthreads();
  const int ty = tid >> 4, tx = tid & 15;
  for (int oc = 0; oc < 48; ++oc) {
    float acc = bias[oc];
    for (int ci = 0; ci < 12; ++ci) {
      const float* ip = &sIn[ci * 324 + ty * 18 + tx];
      const float* wp = &sW[(oc * 12 + ci) * 9];
#pragma unroll
      for (int ky = 0; ky < 3; ++ky)
#pragma unroll
        for (int kx = 0; kx < 3; ++kx)
          acc += ip[ky * 18 + kx] * wp[ky * 3 + kx];
    }
    float r = fmaxf(acc, 0.f);
    dst[(((size_t)b * 48 + oc) * FEATD + h0 + ty) * WDIM + w0 + tx] = (bf16_t)(r * r);
  }
}

// ------------------------------- conv2 3x3 (48->12) + residual + final writes
__global__ __launch_bounds__(256) void k_conv2(
    const bf16_t* __restrict__ src, const float* __restrict__ w,
    const float* __restrict__ bias, const float* __restrict__ h1,
    float* __restrict__ out0, float* __restrict__ outh) {
  __shared__ bf16_t sIn[48 * 18 * 18];
  __shared__ float sW[12 * 48 * 9];
  const int b = blockIdx.z;
  const int h0 = blockIdx.y * 16, w0 = blockIdx.x * 16;
  const int tid = threadIdx.x;
  for (int li = tid; li < 12 * 48 * 9; li += 256) sW[li] = w[li];
  for (int li = tid; li < 48 * 324; li += 256) {
    int ci = li / 324, rem = li % 324;
    int iy = rem / 18, ix = rem % 18;
    int h = h0 + iy - 1, ww = w0 + ix - 1;
    bf16_t v = (bf16_t)0.f;
    if (h >= 0 && h < FEATD && ww >= 0 && ww < WDIM)
      v = src[(((size_t)b * 48 + ci) * FEATD + h) * WDIM + ww];
    sIn[li] = v;
  }
  __syncthreads();
  const int ty = tid >> 4, tx = tid & 15;
  for (int oc = 0; oc < CH; ++oc) {
    float acc = bias[oc];
    for (int ci = 0; ci < 48; ++ci) {
      const bf16_t* ip = &sIn[ci * 324 + ty * 18 + tx];
      const float* wp = &sW[(oc * 48 + ci) * 9];
#pragma unroll
      for (int ky = 0; ky < 3; ++ky)
#pragma unroll
        for (int kx = 0; kx < 3; ++kx)
          acc += (float)ip[ky * 18 + kx] * wp[ky * 3 + kx];
    }
    size_t sp = ((size_t)h0 + ty) * WDIM + w0 + tx;
    float hf = h1[((size_t)b * CH + oc) * FEATD * WDIM + sp] + acc;
    out0[((size_t)(b * 15 + 3 + oc)) * FEATD * WDIM + sp] = hf;  // concat ch 3..14
    outh[((size_t)b * CH + oc) * FEATD * WDIM + sp] = hf;        // h output
  }
}

// ============================================================== launcher
extern "C" void kernel_launch(void* const* d_in, const int* in_sizes, int n_in,
                              void* d_out, int out_size, void* d_ws, size_t ws_size,
                              hipStream_t stream) {
  (void)in_sizes; (void)n_in; (void)out_size; (void)ws_size;
  const float* x    = (const float*)d_in[0];
  const float* ew   = (const float*)d_in[1];
  const float* eb   = (const float*)d_in[2];
  const float* ln1w = (const float*)d_in[3];
  const float* ln1b = (const float*)d_in[4];
  const float* q_dw = (const float*)d_in[5];
  const float* q_pw = (const float*)d_in[6];
  const float* k_dw = (const float*)d_in[7];
  const float* k_pw = (const float*)d_in[8];
  const float* v_dw = (const float*)d_in[9];
  const float* v_pw = (const float*)d_in[10];
  const float* o_pw = (const float*)d_in[11];
  const float* ln2w = (const float*)d_in[12];
  const float* ln2b = (const float*)d_in[13];
  const float* c1w  = (const float*)d_in[14];
  const float* c1b  = (const float*)d_in[15];
  const float* c2w  = (const float*)d_in[16];
  const float* c2b  = (const float*)d_in[17];

  float* out = (float*)d_out;
  const size_t NE = (size_t)BDIM * CH * FEATD * WDIM;       // 12,582,912
  const size_t o_h  = (size_t)BDIM * 15 * FEATD * WDIM;     // concat size
  const size_t o_qk = o_h + NE;                             // qk offset

  // workspace layout (bytes); total ~529 MB
  char* ws = (char*)d_ws;
  float*  h0   = (float*)(ws + 0);
  float*  h1   = (float*)(ws + 50331648ull);
  float*  lnb_ = (float*)(ws + 100663296ull);
  float*  ybuf = (float*)(ws + 150994944ull);
  bf16_t* dwb  = (bf16_t*)(ws + 201326592ull);
  bf16_t* wq   = (bf16_t*)(ws + 226492416ull);
  bf16_t* wk   = (bf16_t*)(ws + 251658240ull);
  bf16_t* wv   = (bf16_t*)(ws + 276824064ull);
  bf16_t* wo   = (bf16_t*)(ws + 301989888ull);
  bf16_t* qr   = (bf16_t*)(ws + 327155712ull);
  bf16_t* kr   = (bf16_t*)(ws + 352321536ull);
  bf16_t* vb   = (bf16_t*)(ws + 377487360ull);
  bf16_t* attb = (bf16_t*)(ws + 402653184ull);
  bf16_t* tbuf = (bf16_t*)(ws + 427819008ull);

  const long long wn = (long long)CH * FEATD * FEATD;
  const dim3 blk(256);
  const dim3 gG(WDIM / 32, FEATD / 64, BDIM * CH);   // WMMA GEMM grid
  const dim3 gDW(WDIM / 16, FEATD / 16, BDIM * CH);  // depthwise grid
  const dim3 gC(WDIM / 16, FEATD / 16, BDIM);        // 3x3 conv grid
  const dim3 gA(WDIM / 16, BDIM * CH * HEADS);       // attention grid
  const size_t gemmLds = (size_t)(64 * 40 + 32 * 40) * sizeof(bf16_t); // 7680 B

  k_copy_x<<<(unsigned)((BDIM * 3 * (size_t)FEATD * WDIM + 255) / 256), blk, 0, stream>>>(x, out);
  k_embed<<<(unsigned)((NE + 255) / 256), blk, 0, stream>>>(x, ew, eb, h0);
  k_layernorm<<<BDIM * CH * WDIM, blk, 0, stream>>>(h0, lnb_, ln1w, ln1b);

  k_f32_to_bf16<<<(unsigned)((wn + 255) / 256), blk, 0, stream>>>(q_pw, wq, wn);
  k_f32_to_bf16<<<(unsigned)((wn + 255) / 256), blk, 0, stream>>>(k_pw, wk, wn);
  k_f32_to_bf16<<<(unsigned)((wn + 255) / 256), blk, 0, stream>>>(v_pw, wv, wn);
  k_f32_to_bf16<<<(unsigned)((wn + 255) / 256), blk, 0, stream>>>(o_pw, wo, wn);

  const int ropeN = BDIM * CH * HEADS * WDIM * (HDIM / 2);
  // q path
  k_dwconv9<<<gDW, blk, 0, stream>>>(lnb_, q_dw, dwb);
  k_gemm_bf16<<<gG, blk, gemmLds, stream>>>(wq, dwb, ybuf, FEATD, WDIM, FEATD,
                                            (long long)FEATD * FEATD, CH);
  k_rope_pack<<<(ropeN + 255) / 256, blk, 0, stream>>>(ybuf, qr);
  // k path
  k_dwconv9<<<gDW, blk, 0, stream>>>(lnb_, k_dw, dwb);
  k_gemm_bf16<<<gG, blk, gemmLds, stream>>>(wk, dwb, ybuf, FEATD, WDIM, FEATD,
                                            (long long)FEATD * FEATD, CH);
  k_rope_pack<<<(ropeN + 255) / 256, blk, 0, stream>>>(ybuf, kr);
  // v path
  k_dwconv9<<<gDW, blk, 0, stream>>>(lnb_, v_dw, dwb);
  k_gemm_bf16<<<gG, blk, gemmLds, stream>>>(wv, dwb, ybuf, FEATD, WDIM, FEATD,
                                            (long long)FEATD * FEATD, CH);
  k_pack_v<<<(unsigned)((NE + 255) / 256), blk, 0, stream>>>(ybuf, vb);

  // fused attention: writes qk output + attention result (bf16, (b,c,f,w))
  k_attention<<<gA, blk, 0, stream>>>(qr, kr, vb, out + o_qk, attb);

  // output projection + residual
  k_gemm_bf16<<<gG, blk, gemmLds, stream>>>(wo, attb, ybuf, FEATD, WDIM, FEATD,
                                            (long long)FEATD * FEATD, CH);
  k_add<<<(unsigned)((NE + 255) / 256), blk, 0, stream>>>(h0, ybuf, h1, (long long)NE);

  // FFN
  k_layernorm<<<BDIM * CH * WDIM, blk, 0, stream>>>(h1, lnb_, ln2w, ln2b);
  k_conv1<<<gC, blk, 0, stream>>>(lnb_, c1w, c1b, tbuf);
  k_conv2<<<gC, blk, 0, stream>>>(tbuf, c2w, c2b, h1, out, out + o_h);
}
